// CausalSelfAttention_84928683311445
// MI455X (gfx1250) — matrix operand order repaired and checked by reading, not compiled, not run on hardware
//
#include <hip/hip_runtime.h>
#include <hip/hip_bf16.h>
#include <stdint.h>

// Problem constants (match reference)
#define TT 2048
#define CC 1024
#define HH 16
#define HSZ 64
#define BBATCH 4
#define BT (BBATCH * TT)   // 8192 rows

typedef __attribute__((ext_vector_type(16))) __bf16 v16bf;
typedef __attribute__((ext_vector_type(8)))  float  v8f;

struct Frag { union { v16bf v; uint4 q[2]; }; };

__device__ __forceinline__ uint16_t f32_to_bf16(float f) {
    uint32_t u = __float_as_uint(f);
    u += 0x7FFFu + ((u >> 16) & 1u);      // round-to-nearest-even
    return (uint16_t)(u >> 16);
}

__device__ __forceinline__ v8f wmma_bf16(const Frag& a, const Frag& b, v8f c) {
    return __builtin_amdgcn_wmma_f32_16x16x32_bf16(
        /*neg_a=*/false, a.v, /*neg_b=*/false, b.v,
        /*c_mod=*/(short)0, c, /*reuse_a=*/false, /*reuse_b=*/false);
}

// generic shared-memory pointer -> LDS byte offset (low 32 bits of aperture addr)
__device__ __forceinline__ uint32_t lds_off(const void* p) {
    return (uint32_t)(uintptr_t)p;
}

// async global -> LDS, 16 bytes per lane, tracked by ASYNCcnt
__device__ __forceinline__ void async_b128(uint32_t ldsaddr, const void* gptr) {
    unsigned long long ga = (unsigned long long)(uintptr_t)gptr;
    asm volatile("global_load_async_to_lds_b128 %0, %1, off"
                 :: "v"(ldsaddr), "v"(ga) : "memory");
}
__device__ __forceinline__ void wait_async0() {
    asm volatile("s_wait_asynccnt 0x0" ::: "memory");
}

// ---- 16-lane xor butterfly via v_permlane16_b32 (VALU, no LDS round-trip) ----
constexpr uint32_t psel_lo(int M) {
    uint32_t r = 0;
    for (int i = 0; i < 8; ++i) r |= (uint32_t)((i ^ M) & 0xF) << (4 * i);
    return r;
}
constexpr uint32_t psel_hi(int M) {
    uint32_t r = 0;
    for (int i = 8; i < 16; ++i) r |= (uint32_t)((i ^ M) & 0xF) << (4 * (i - 8));
    return r;
}
#if __has_builtin(__builtin_amdgcn_permlane16)
template <int M>
__device__ __forceinline__ float lane_xor16(float x) {
    constexpr uint32_t s0 = psel_lo(M), s1 = psel_hi(M);
    unsigned v = __float_as_uint(x);
    unsigned r = __builtin_amdgcn_permlane16(v, v, s0, s1, false, false);
    return __uint_as_float(r);
}
#else
template <int M>
__device__ __forceinline__ float lane_xor16(float x) {
    return __shfl_xor(x, M, 32);
}
#endif
__device__ __forceinline__ float redmax16(float t) {
    t = fmaxf(t, lane_xor16<1>(t));
    t = fmaxf(t, lane_xor16<2>(t));
    t = fmaxf(t, lane_xor16<4>(t));
    t = fmaxf(t, lane_xor16<8>(t));
    return t;
}
__device__ __forceinline__ float redsum16(float t) {
    t += lane_xor16<1>(t);
    t += lane_xor16<2>(t);
    t += lane_xor16<4>(t);
    t += lane_xor16<8>(t);
    return t;
}

// ---------------------------------------------------------------- prep kernels
__global__ void cvt_f32_bf16(const float* __restrict__ in,
                             uint16_t* __restrict__ out, int n) {
    int i = (blockIdx.x * blockDim.x + threadIdx.x) * 4;
    if (i + 3 < n) {
        float4 f = *(const float4*)(in + i);
        ushort4 o;
        o.x = f32_to_bf16(f.x); o.y = f32_to_bf16(f.y);
        o.z = f32_to_bf16(f.z); o.w = f32_to_bf16(f.w);
        *(ushort4*)(out + i) = o;
    }
}

// in: [K][N] f32 row-major -> out: [N][K] bf16, LDS-tiled 32x32 transpose
__global__ __launch_bounds__(256)
void transpose_f32_bf16(const float* __restrict__ in,
                        uint16_t* __restrict__ out, int K, int N) {
    __shared__ float tile[32][33];
    const int bk = blockIdx.x * 32;   // K tile base
    const int bn = blockIdx.y * 32;   // N tile base
    const int tx = threadIdx.x & 31;
    const int ty = threadIdx.x >> 5;  // 0..7
    #pragma unroll
    for (int r = 0; r < 32; r += 8)
        tile[ty + r][tx] = in[(size_t)(bk + ty + r) * N + bn + tx];
    __syncthreads();
    #pragma unroll
    for (int r = 0; r < 32; r += 8)
        out[(size_t)(bn + ty + r) * K + bk + tx] = f32_to_bf16(tile[tx][ty + r]);
}

// ---------------------------------------------------------------- GEMM
// C[M][N] = A[M][K] * Bt[N][K]^T + bias[N]
// block tile 128x256, 8 waves (2x4), wave tile 64x64, double-buffered async LDS.
template <bool OUT_F32>
__global__ __launch_bounds__(256)
void gemm_bf16(const uint16_t* __restrict__ A, const uint16_t* __restrict__ Bt,
               const float* __restrict__ bias, void* __restrict__ Cout,
               int M, int N, int K) {
    __shared__ __align__(16) uint16_t As[2][128 * 40];   // 32 K + 8 pad
    __shared__ __align__(16) uint16_t Bs[2][256 * 40];

    const int tid  = threadIdx.x;
    const int wave = tid >> 5;
    const int lane = tid & 31;
    const int lh   = lane >> 4;
    const int lr   = lane & 15;
    const int m0 = blockIdx.y * 128;
    const int n0 = blockIdx.x * 256;
    const int wm = wave >> 2;     // 0..1
    const int wn = wave & 3;      // 0..3

    // staging: A -> thread loads half a row (16 elems), B -> thread loads a full row (32)
    const int arow = tid >> 1, acol = (tid & 1) * 16;
    const uint16_t* Ab = A  + (size_t)m0 * K + (size_t)arow * K + acol;
    const uint16_t* Bb = Bt + (size_t)n0 * K + (size_t)tid * K;
    const uint32_t aoff = (uint32_t)(arow * 40 + acol);
    const uint32_t boff = (uint32_t)(tid * 40);

    v8f acc[4][4] = {};

    // prologue: stage tile 0 into buffer 0
    async_b128(lds_off(&As[0][aoff]),     Ab);
    async_b128(lds_off(&As[0][aoff + 8]), Ab + 8);
    #pragma unroll
    for (int c = 0; c < 4; ++c)
        async_b128(lds_off(&Bs[0][boff + c * 8]), Bb + c * 8);

    const int nK = K >> 5;
    for (int kt = 0; kt < nK; ++kt) {
        const int buf = kt & 1;
        wait_async0();          // own asyncs for buf done
        __syncthreads();        // everyone's asyncs done; prev compute done
        if (kt + 1 < nK) {      // stream next tile while we compute
            const int k1 = (kt + 1) << 5;
            const int nb = buf ^ 1;
            async_b128(lds_off(&As[nb][aoff]),     Ab + k1);
            async_b128(lds_off(&As[nb][aoff + 8]), Ab + k1 + 8);
            #pragma unroll
            for (int c = 0; c < 4; ++c)
                async_b128(lds_off(&Bs[nb][boff + c * 8]), Bb + k1 + c * 8);
        }

        Frag af[4], bq[4];
        #pragma unroll
        for (int mi = 0; mi < 4; ++mi) {
            const uint16_t* p = &As[buf][(wm * 64 + mi * 16 + lr) * 40 + lh * 8];
            af[mi].q[0] = *(const uint4*)(p);
            af[mi].q[1] = *(const uint4*)(p + 16);
        }
        #pragma unroll
        for (int ni = 0; ni < 4; ++ni) {
            const uint16_t* p = &Bs[buf][(wn * 64 + ni * 16 + lr) * 40 + lh * 8];
            bq[ni].q[0] = *(const uint4*)(p);
            bq[ni].q[1] = *(const uint4*)(p + 16);
        }
        #pragma unroll
        for (int mi = 0; mi < 4; ++mi)
            #pragma unroll
            for (int ni = 0; ni < 4; ++ni)
                acc[mi][ni] = wmma_bf16(af[mi], bq[ni], acc[mi][ni]);
    }

    float bv[4];
    #pragma unroll
    for (int ni = 0; ni < 4; ++ni)
        bv[ni] = bias[n0 + wn * 64 + ni * 16 + lr];

    #pragma unroll
    for (int mi = 0; mi < 4; ++mi)
        #pragma unroll
        for (int ni = 0; ni < 4; ++ni)
            #pragma unroll
            for (int v = 0; v < 8; ++v) {
                int row = m0 + wm * 64 + mi * 16 + v + 8 * lh;
                int col = n0 + wn * 64 + ni * 16 + lr;
                float r = acc[mi][ni][v] + bv[ni];
                if (OUT_F32)
                    ((float*)Cout)[(size_t)row * N + col] = r;
                else
                    ((uint16_t*)Cout)[(size_t)row * N + col] = f32_to_bf16(r);
            }
}

// ---------------------------------------------------------------- flash attention
// qkv: [B*T][3C] bf16; attout: [B*T][C] bf16
// grid: (T/128, H, B), 256 threads. Br=128 (16 q rows / wave), Bc=64.
// K tile: double-buffered async; V tile: register-prefetched + transpose scatter.
// Softmax kept in base-2 domain (exp2), reductions via v_permlane16.
__global__ __launch_bounds__(256)
void attn_kernel(const uint16_t* __restrict__ qkv, uint16_t* __restrict__ attout) {
    __shared__ __align__(16) uint16_t Kt[2][64 * 72];   // [key][hs] + pad
    __shared__ __align__(16) uint16_t Vt[2][64 * 72];   // [hs][key] + pad (transposed)
    __shared__ __align__(16) uint16_t Pt[128 * 72];     // [q][key]  + pad

    const int qi = blockIdx.x;
    const int h  = blockIdx.y;
    const int b  = blockIdx.z;
    const int tid  = threadIdx.x;
    const int wave = tid >> 5;
    const int lane = tid & 31;
    const int lh   = lane >> 4;
    const int lr   = lane & 15;

    const int rowStride = 3 * CC;
    const uint16_t* qbase = qkv + (size_t)b * TT * rowStride + h * HSZ;
    const uint16_t* kbase = qbase + CC;
    const uint16_t* vbase = qbase + 2 * CC;

    const int q0 = qi * 128;
    // Q A-fragments (row = q0 + 16*wave + lr), hs split into two K=32 steps
    Frag qf[2];
    {
        const uint16_t* qp = qbase + (size_t)(q0 + wave * 16 + lr) * rowStride + lh * 8;
        qf[0].q[0] = *(const uint4*)(qp);
        qf[0].q[1] = *(const uint4*)(qp + 16);
        qf[1].q[0] = *(const uint4*)(qp + 32);
        qf[1].q[1] = *(const uint4*)(qp + 48);
    }

    float m_i[8], l_i[8];
    v8f o[4] = {};
    #pragma unroll
    for (int v = 0; v < 8; ++v) { m_i[v] = -__builtin_inff(); l_i[v] = 0.0f; }

    // base-2 softmax: scale = (1/sqrt(64)) * log2(e)
    const float scale2 = 0.125f * 1.44269504088896340736f;
    const int jmax = 2 * qi + 1; // last key block touching q0+127 (causal)

    // per-thread staging chunks for 64x64 bf16 tile (2 x 16B per thread)
    const int kr0 = (tid * 2) >> 3,     kc0 = ((tid * 2) & 7) * 8;
    const int kr1 = (tid * 2 + 1) >> 3, kc1 = ((tid * 2 + 1) & 7) * 8;

    // prologue: async K(0) into Kt[0]; V(0) into registers
    async_b128(lds_off(&Kt[0][kr0 * 72 + kc0]), kbase + (size_t)kr0 * rowStride + kc0);
    async_b128(lds_off(&Kt[0][kr1 * 72 + kc1]), kbase + (size_t)kr1 * rowStride + kc1);
    uint4 vr0 = *(const uint4*)(vbase + (size_t)kr0 * rowStride + kc0);
    uint4 vr1 = *(const uint4*)(vbase + (size_t)kr1 * rowStride + kc1);

    for (int j = 0; j <= jmax; ++j) {
        const int buf = j & 1;
        wait_async0();          // own K(j) asyncs done
        __syncthreads();        // everyone's K(j) done; compute(j-1) done

        // scatter current V registers into Vt[buf] (transpose to [hs][key])
        {
            const uint16_t* pe = (const uint16_t*)&vr0;
            #pragma unroll
            for (int e = 0; e < 8; ++e) Vt[buf][(kc0 + e) * 72 + kr0] = pe[e];
            pe = (const uint16_t*)&vr1;
            #pragma unroll
            for (int e = 0; e < 8; ++e) Vt[buf][(kc1 + e) * 72 + kr1] = pe[e];
        }
        // prefetch next tile: async K(j+1); V(j+1) into registers
        if (j < jmax) {
            const int kb1 = (j + 1) * 64;
            async_b128(lds_off(&Kt[buf ^ 1][kr0 * 72 + kc0]),
                       kbase + (size_t)(kb1 + kr0) * rowStride + kc0);
            async_b128(lds_off(&Kt[buf ^ 1][kr1 * 72 + kc1]),
                       kbase + (size_t)(kb1 + kr1) * rowStride + kc1);
            vr0 = *(const uint4*)(vbase + (size_t)(kb1 + kr0) * rowStride + kc0);
            vr1 = *(const uint4*)(vbase + (size_t)(kb1 + kr1) * rowStride + kc1);
        }
        __syncthreads();        // Vt[buf] scatters visible to all waves

        const int kblk = j * 64;

        // S = Q K^T : 4 key-fragments of 16, two K=32 WMMA steps over hs
        v8f s[4];
        #pragma unroll
        for (int nf = 0; nf < 4; ++nf) {
            const uint16_t* kp = &Kt[buf][(16 * nf + lr) * 72 + lh * 8];
            Frag b0, b1;
            b0.q[0] = *(const uint4*)(kp);      b0.q[1] = *(const uint4*)(kp + 16);
            b1.q[0] = *(const uint4*)(kp + 32); b1.q[1] = *(const uint4*)(kp + 48);
            v8f z = {};
            z = wmma_bf16(qf[0], b0, z);
            z = wmma_bf16(qf[1], b1, z);
            s[nf] = z;
        }

        // scale (log2 domain) + causal mask (C layout: row = v + 8*lh, col = 16*nf + lr)
        #pragma unroll
        for (int nf = 0; nf < 4; ++nf)
            #pragma unroll
            for (int v = 0; v < 8; ++v) {
                int qr = q0 + wave * 16 + v + 8 * lh;
                int kc = kblk + 16 * nf + lr;
                float val = s[nf][v] * scale2;
                s[nf][v] = (kc > qr) ? -__builtin_inff() : val;
            }

        // online softmax: row reductions across 16 lanes of each half (permlane16)
        float alpha[8];
        #pragma unroll
        for (int v = 0; v < 8; ++v) {
            float t = fmaxf(fmaxf(s[0][v], s[1][v]), fmaxf(s[2][v], s[3][v]));
            t = redmax16(t);
            float mn = fmaxf(m_i[v], t);
            alpha[v] = exp2f(m_i[v] - mn);
            m_i[v] = mn;
        }
        #pragma unroll
        for (int v = 0; v < 8; ++v) {
            float rs = 0.0f;
            #pragma unroll
            for (int nf = 0; nf < 4; ++nf) {
                float p = exp2f(s[nf][v] - m_i[v]);
                s[nf][v] = p;
                rs += p;
            }
            rs = redsum16(rs);
            l_i[v] = l_i[v] * alpha[v] + rs;
        }
        #pragma unroll
        for (int nf = 0; nf < 4; ++nf)
            #pragma unroll
            for (int v = 0; v < 8; ++v)
                o[nf][v] *= alpha[v];

        // P (C layout f32) -> LDS bf16, per-wave private rows
        #pragma unroll
        for (int nf = 0; nf < 4; ++nf)
            #pragma unroll
            for (int v = 0; v < 8; ++v)
                Pt[(wave * 16 + v + 8 * lh) * 72 + 16 * nf + lr] = f32_to_bf16(s[nf][v]);

        asm volatile("s_wait_dscnt 0x0" ::: "memory");

        // reload P as A-fragments (two K=32 steps over 64 keys)
        Frag pf[2];
        {
            const uint16_t* pp = &Pt[(wave * 16 + lr) * 72 + lh * 8];
            pf[0].q[0] = *(const uint4*)(pp);
            pf[0].q[1] = *(const uint4*)(pp + 16);
            pf[1].q[0] = *(const uint4*)(pp + 32);
            pf[1].q[1] = *(const uint4*)(pp + 48);
        }
        // O += P * V ; B-fragments from Vt [hs][key]
        #pragma unroll
        for (int nf = 0; nf < 4; ++nf) {
            const uint16_t* vp = &Vt[buf][(16 * nf + lr) * 72 + lh * 8];
            Frag vb0, vb1;
            vb0.q[0] = *(const uint4*)(vp);      vb0.q[1] = *(const uint4*)(vp + 16);
            vb1.q[0] = *(const uint4*)(vp + 32); vb1.q[1] = *(const uint4*)(vp + 48);
            o[nf] = wmma_bf16(pf[0], vb0, o[nf]);
            o[nf] = wmma_bf16(pf[1], vb1, o[nf]);
        }
    }

    // normalize and store (bf16) to attout [B*T][C]
    float inv_l[8];
    #pragma unroll
    for (int v = 0; v < 8; ++v) inv_l[v] = 1.0f / l_i[v];
    #pragma unroll
    for (int nf = 0; nf < 4; ++nf)
        #pragma unroll
        for (int v = 0; v < 8; ++v) {
            int row = q0 + wave * 16 + v + 8 * lh;
            int col = h * HSZ + 16 * nf + lr;
            attout[(size_t)(b * TT + row) * CC + col] = f32_to_bf16(o[nf][v] * inv_l[v]);
        }
}

// ---------------------------------------------------------------- launch
extern "C" void kernel_launch(void* const* d_in, const int* in_sizes, int n_in,
                              void* d_out, int out_size, void* d_ws, size_t ws_size,
                              hipStream_t stream) {
    const float* x      = (const float*)d_in[0];   // [B,T,C]
    const float* w_attn = (const float*)d_in[1];   // [C,3C]
    const float* b_attn = (const float*)d_in[2];   // [3C]
    const float* w_proj = (const float*)d_in[3];   // [C,C]
    const float* b_proj = (const float*)d_in[4];   // [C]
    float* out = (float*)d_out;

    char* ws = (char*)d_ws;
    uint16_t* xb   = (uint16_t*)ws;  ws += (size_t)BT * CC * 2;      // 16 MB
    uint16_t* waT  = (uint16_t*)ws;  ws += (size_t)3 * CC * CC * 2;  //  6 MB
    uint16_t* wpT  = (uint16_t*)ws;  ws += (size_t)CC * CC * 2;      //  2 MB
    uint16_t* qkvb = (uint16_t*)ws;  ws += (size_t)BT * 3 * CC * 2;  // 48 MB
    uint16_t* attb = (uint16_t*)ws;                                   // 16 MB

    // 1) precision conversion + weight transposition (bf16, [N][K])
    {
        int n = BT * CC;
        cvt_f32_bf16<<<n / (256 * 4), 256, 0, stream>>>(x, xb, n);
        transpose_f32_bf16<<<dim3(CC / 32, 3 * CC / 32), 256, 0, stream>>>(w_attn, waT, CC, 3 * CC);
        transpose_f32_bf16<<<dim3(CC / 32, CC / 32), 256, 0, stream>>>(w_proj, wpT, CC, CC);
    }

    // 2) QKV projection: [8192,1024] x [1024,3072] -> bf16 qkv
    gemm_bf16<false><<<dim3(3 * CC / 256, BT / 128), 256, 0, stream>>>(
        xb, waT, b_attn, (void*)qkvb, BT, 3 * CC, CC);

    // 3) flash attention -> bf16 [B*T][C]
    attn_kernel<<<dim3(TT / 128, HH, BBATCH), 256, 0, stream>>>(qkvb, attb);

    // 4) output projection: [8192,1024] x [1024,1024] -> f32 out
    gemm_bf16<true><<<dim3(CC / 256, BT / 128), 256, 0, stream>>>(
        attb, wpT, b_proj, (void*)out, BT, CC, CC);
}
